// GPTDecoder_44650480009687
// MI455X (gfx1250) — matrix operand order repaired and checked by reading, not compiled, not run on hardware
//
#include <hip/hip_runtime.h>
#include <hip/hip_bf16.h>
#include <math.h>

// ---- problem constants (match reference) ----
#define V_    32000
#define S_    1024
#define D_    768
#define H_    12
#define HID_  3072
#define L_    4
#define B_    2
#define HD_   64
#define M_    (B_ * S_)     // 2048 token rows
#define EPS_  1e-5f

typedef _Float16 v16h __attribute__((ext_vector_type(16)));
typedef _Float16 h8   __attribute__((ext_vector_type(8)));
typedef float    v8f  __attribute__((ext_vector_type(8)));

// ----------------------------------------------------------------------------
// CDNA5 async global->LDS copy (ASYNCcnt-tracked, bypasses VGPRs)
// ----------------------------------------------------------------------------
__device__ __forceinline__ void async_b128(unsigned lds_off,
                                           const _Float16* gptr) {
  asm volatile("global_load_async_to_lds_b128 %0, %1, off"
               :: "v"(lds_off), "v"(gptr) : "memory");
}
__device__ __forceinline__ void wait_async_10() {
  asm volatile("s_wait_asynccnt 10" ::: "memory");
}
__device__ __forceinline__ void wait_async_0() {
  asm volatile("s_wait_asynccnt 0" ::: "memory");
}

// ----------------------------------------------------------------------------
// f32 -> f16 conversion (weights), grid-stride
// ----------------------------------------------------------------------------
__global__ void cvt_f32_f16(const float* __restrict__ in,
                            _Float16* __restrict__ out, int n) {
  int i = blockIdx.x * blockDim.x + threadIdx.x;
  int stride = gridDim.x * blockDim.x;
  for (; i < n; i += stride) out[i] = (_Float16)in[i];
}

// ----------------------------------------------------------------------------
// Embedding gather: x[b,s,:] = W_emb[id]*sqrt(D) + pos_emb[s]
// ----------------------------------------------------------------------------
__global__ void embed_kernel(const float* __restrict__ W_emb,
                             const float* __restrict__ pos_emb,
                             const int* __restrict__ ids,
                             float* __restrict__ x) {
  const int row = blockIdx.x;           // b*S + s
  const int s = row & (S_ - 1);
  const int id = ids[row];
  const float* w = W_emb + (long long)id * D_;
  const float* p = pos_emb + (long long)s * D_;
  float* xr = x + (long long)row * D_;
  for (int j = threadIdx.x; j < D_; j += blockDim.x)
    xr[j] = w[j] * 27.712812921102035f + p[j];   // sqrt(768)
}

// ----------------------------------------------------------------------------
// LayerNorm (fp32 in) -> f16 out; one 256-thread block per row (D=768 = 3/thr)
// ----------------------------------------------------------------------------
__global__ __launch_bounds__(256) void layernorm_f16(
    const float* __restrict__ x, const float* __restrict__ g,
    const float* __restrict__ b, _Float16* __restrict__ out) {
  __shared__ float red[256];
  const int row = blockIdx.x;
  const int t = threadIdx.x;
  const float* xr = x + (long long)row * D_;
  float v0 = xr[t], v1 = xr[t + 256], v2 = xr[t + 512];

  red[t] = v0 + v1 + v2;
  __syncthreads();
  for (int o = 128; o > 0; o >>= 1) {
    if (t < o) red[t] += red[t + o];
    __syncthreads();
  }
  const float mean = red[0] * (1.0f / D_);
  __syncthreads();

  const float d0 = v0 - mean, d1 = v1 - mean, d2 = v2 - mean;
  red[t] = d0 * d0 + d1 * d1 + d2 * d2;
  __syncthreads();
  for (int o = 128; o > 0; o >>= 1) {
    if (t < o) red[t] += red[t + o];
    __syncthreads();
  }
  const float rstd = rsqrtf(red[0] * (1.0f / D_) + EPS_);

  _Float16* orow = out + (long long)row * D_;
  orow[t]       = (_Float16)(g[t]       * (d0 * rstd) + b[t]);
  orow[t + 256] = (_Float16)(g[t + 256] * (d1 * rstd) + b[t + 256]);
  orow[t + 512] = (_Float16)(g[t + 512] * (d2 * rstd) + b[t + 512]);
}

// ----------------------------------------------------------------------------
// WMMA GEMM: acc[M,N] = A[M,K](f16) @ W[N,K](f16)^T + bias (+ residual)
//   Cf32 (optional): f32 store of result (ntstore=1 -> non-temporal)
//   Cf16 (optional): f16 store of result (after optional exact-GELU)
// block = 256 thr (8 wave32), block tile 64x256, wave tile 32x64,
// per-wave 2x4 grid of v_wmma_f32_16x16x32_f16 accumulators.
// K staged through double-buffered LDS in 64-wide slabs via
// global_load_async_to_lds_b128 (ASYNCcnt); 8 lanes cover one row's 128B so
// each 8-lane group issues contiguous cacheline requests.
// Requires M%64==0, N%256==0, K%64==0, K>=128.
// ----------------------------------------------------------------------------
__global__ __launch_bounds__(256) void gemm_wmma_f16(
    const _Float16* __restrict__ A, const _Float16* __restrict__ W,
    const float* __restrict__ bias, const float* __restrict__ residual,
    float* __restrict__ Cf32, _Float16* __restrict__ Cf16, int gelu,
    int ntstore, int M, int N, int K) {
  __shared__ __align__(32) _Float16 sA[2][64 * 64];    //  2 x  8 KB
  __shared__ __align__(32) _Float16 sB[2][256 * 64];   //  2 x 32 KB

  const int tid  = threadIdx.x;
  const int lane = tid & 31;
  const int wave = tid >> 5;
  const int wm   = wave >> 2;           // 0..1 (M dir)
  const int wn   = wave & 3;            // 0..3 (N dir)
  const int lhalf = lane >> 4;          // 0/1
  const int lmod  = lane & 15;

  const int blockN = blockIdx.x * 256;
  const int blockM = blockIdx.y * 64;

  v8f acc[2][4] = {};

  // ---- async staging (coalesced): 8 lanes x 16B = one row's 128B slab ----
  // thread t -> row (t>>3) + pass*32, 16B chunk (t&7)
  const int srow = tid >> 3, schk = tid & 7;
  const _Float16* Ag = A + (long long)(blockM + srow) * K + schk * 8;
  const _Float16* Wg = W + (long long)(blockN + srow) * K + schk * 8;
  const unsigned aMy =
      (unsigned)(unsigned long long)&sA[0][srow * 64 + schk * 8];
  const unsigned bMy =
      (unsigned)(unsigned long long)&sB[0][srow * 64 + schk * 8];
  const unsigned aStride = 64 * 64 * sizeof(_Float16);    // 8192
  const unsigned bStride = 256 * 64 * sizeof(_Float16);   // 32768
  const unsigned passB = 32 * 64 * sizeof(_Float16);      // 4096 (32 rows)

  auto issue_slab = [&](int buf, int k0) {
    const unsigned ao = aMy + (unsigned)buf * aStride;
    const unsigned bo = bMy + (unsigned)buf * bStride;
#pragma unroll
    for (int p = 0; p < 2; ++p)                       // A: 64 rows
      async_b128(ao + p * passB, Ag + k0 + (long long)p * 32 * K);
#pragma unroll
    for (int p = 0; p < 8; ++p)                       // B: 256 rows
      async_b128(bo + p * passB, Wg + k0 + (long long)p * 32 * K);
  };

  auto compute_slab = [&](int buf) {
#pragma unroll
    for (int kk = 0; kk < 64; kk += 32) {
      // A fragments: lane holds row M=lane%16; lanes<16 K={0..15},
      // lanes>=16 K={16..31}  (contiguous K, matches B)
      v16h afr[2];
#pragma unroll
      for (int mi = 0; mi < 2; ++mi) {
        const int r = wm * 32 + mi * 16 + lmod;
        h8 lo = *(const h8*)&sA[buf][r * 64 + kk + lhalf * 16];
        h8 hi = *(const h8*)&sA[buf][r * 64 + kk + lhalf * 16 + 8];
        afr[mi] = __builtin_shufflevector(lo, hi, 0, 1, 2, 3, 4, 5, 6, 7,
                                          8, 9, 10, 11, 12, 13, 14, 15);
      }
      // B fragments: lane holds column N=lane%16, 16 contiguous K at
      // 16*(lane/16) — contiguous along K of W's row (W is N x K row-major)
      v16h bfr[4];
#pragma unroll
      for (int ni = 0; ni < 4; ++ni) {
        const int c = wn * 64 + ni * 16 + lmod;
        bfr[ni] = *(const v16h*)&sB[buf][c * 64 + kk + lhalf * 16];
      }

#pragma unroll
      for (int mi = 0; mi < 2; ++mi)
#pragma unroll
        for (int ni = 0; ni < 4; ++ni)
          acc[mi][ni] = __builtin_amdgcn_wmma_f32_16x16x32_f16(
              false, afr[mi], false, bfr[ni], (short)0, acc[mi][ni],
              false, false);
    }
  };

  const int nslab = K >> 6;
  issue_slab(0, 0);

  // steady state: straight-line issue -> wait(10) -> barrier -> 16 wmma -> barrier
  for (int sIdx = 0; sIdx + 1 < nslab; ++sIdx) {
    const int buf = sIdx & 1;
    issue_slab(buf ^ 1, (sIdx + 1) << 6);
    wait_async_10();                  // this wave's current slab landed
    __syncthreads();                  // all waves' slab data visible
    compute_slab(buf);
    __syncthreads();                  // buffer free for overwrite
  }
  // peeled final slab
  wait_async_0();
  __syncthreads();
  compute_slab((nslab - 1) & 1);

  // writeback: C layout — VGPR r: lanes0-15 M=r, lanes16-31 M=r+8; N=lane%16
#pragma unroll
  for (int mi = 0; mi < 2; ++mi) {
#pragma unroll
    for (int ni = 0; ni < 4; ++ni) {
      const int colg = blockN + wn * 64 + ni * 16 + lmod;
      const float bv = bias ? bias[colg] : 0.0f;
#pragma unroll
      for (int r = 0; r < 8; ++r) {
        const int rowg = blockM + wm * 32 + mi * 16 + r + (lhalf << 3);
        const long long idx = (long long)rowg * N + colg;
        float v = acc[mi][ni][r] + bv;
        if (residual) v += residual[idx];
        if (Cf32) {
          if (ntstore) __builtin_nontemporal_store(v, &Cf32[idx]);
          else Cf32[idx] = v;
        }
        if (Cf16) {
          float o = gelu ? 0.5f * v * (1.0f + erff(v * 0.70710678118654752f))
                         : v;
          Cf16[idx] = (_Float16)o;
        }
      }
    }
  }
}

// ----------------------------------------------------------------------------
// Causal attention, online softmax. One wave32 per (b, h, q).
// qkv layout: [b*S + s][3*D], per-head slice at h*3*HD: [q(64) | k(64) | v(64)]
// Lanes split HD=64: lane owns d = lane and lane+32. Writes f16 for proj GEMM.
// ----------------------------------------------------------------------------
__global__ __launch_bounds__(256) void attn_kernel(
    const float* __restrict__ qkv, _Float16* __restrict__ out) {
  const int lane = threadIdx.x & 31;
  const int wid = blockIdx.x * (blockDim.x >> 5) + (threadIdx.x >> 5);
  const int q = wid & (S_ - 1);
  const int h = (wid >> 10) % H_;
  const int b = wid / (S_ * H_);

  const float* base = qkv + (long long)(b * S_) * (3 * D_) + h * (3 * HD_);
  const float* qp = base + (long long)q * (3 * D_);
  const float q0 = qp[lane], q1 = qp[lane + 32];

  float m = -3.4e38f, ssum = 0.0f, acc0 = 0.0f, acc1 = 0.0f;
  for (int k = 0; k <= q; ++k) {
    const float* kp = base + (long long)k * (3 * D_) + HD_;
    float part = q0 * kp[lane] + q1 * kp[lane + 32];
#pragma unroll
    for (int o = 16; o > 0; o >>= 1) part += __shfl_xor(part, o, 32);
    float sc = fminf(fmaxf(part * 0.125f, -1e9f), 1e9f);  // /sqrt(64), clamp
    const float mn = fmaxf(m, sc);
    const float scale = __expf(m - mn);
    const float p = __expf(sc - mn);
    const float* vp = kp + HD_;
    acc0 = acc0 * scale + p * vp[lane];
    acc1 = acc1 * scale + p * vp[lane + 32];
    ssum = ssum * scale + p;
    m = mn;
  }
  const float inv = 1.0f / ssum;
  _Float16* op = out + (long long)(b * S_ + q) * D_ + h * HD_;
  op[lane] = (_Float16)(acc0 * inv);
  op[lane + 32] = (_Float16)(acc1 * inv);
}

// ----------------------------------------------------------------------------
// host-side orchestration
// ----------------------------------------------------------------------------
extern "C" void kernel_launch(void* const* d_in, const int* in_sizes, int n_in,
                              void* d_out, int out_size, void* d_ws,
                              size_t ws_size, hipStream_t stream) {
  (void)in_sizes; (void)n_in; (void)out_size; (void)ws_size;

  const float* W_emb   = (const float*)d_in[0];
  const float* pos_emb = (const float*)d_in[1];
  const float* norm1_g = (const float*)d_in[2];
  const float* norm1_b = (const float*)d_in[3];
  const float* qkv_w   = (const float*)d_in[4];
  const float* qkv_b   = (const float*)d_in[5];
  const float* proj_w  = (const float*)d_in[6];
  const float* proj_b  = (const float*)d_in[7];
  const float* norm2_g = (const float*)d_in[8];
  const float* norm2_b = (const float*)d_in[9];
  const float* ffn_w1  = (const float*)d_in[10];
  const float* ffn_b1  = (const float*)d_in[11];
  const float* ffn_w2  = (const float*)d_in[12];
  const float* ffn_b2  = (const float*)d_in[13];
  const float* fin_g   = (const float*)d_in[14];
  const float* fin_b   = (const float*)d_in[15];
  const int*   ids     = (const int*)d_in[16];

  char* ws = (char*)d_ws;
  auto alloc = [&](size_t bytes) -> void* {
    void* p = (void*)ws;
    ws += (bytes + 255) & ~(size_t)255;
    return p;
  };

  float*    x      = (float*)   alloc(sizeof(float)    * M_ * D_);
  _Float16* h16    = (_Float16*)alloc(sizeof(_Float16) * M_ * D_);
  float*    qkvf   = (float*)   alloc(sizeof(float)    * M_ * 3 * D_);
  _Float16* attn16 = (_Float16*)alloc(sizeof(_Float16) * M_ * D_);
  _Float16* hid16  = (_Float16*)alloc(sizeof(_Float16) * M_ * HID_);
  _Float16* wemb16 = (_Float16*)alloc(sizeof(_Float16) * (size_t)V_ * D_);
  _Float16* qkvw16 = (_Float16*)alloc(sizeof(_Float16) * (size_t)L_ * 3 * D_ * D_);
  _Float16* projw16= (_Float16*)alloc(sizeof(_Float16) * (size_t)L_ * D_ * D_);
  _Float16* f1w16  = (_Float16*)alloc(sizeof(_Float16) * (size_t)L_ * HID_ * D_);
  _Float16* f2w16  = (_Float16*)alloc(sizeof(_Float16) * (size_t)L_ * D_ * HID_);

  // one-time (per call) weight precision conversion
  cvt_f32_f16<<<2048, 256, 0, stream>>>(W_emb,  wemb16,  V_ * D_);
  cvt_f32_f16<<<1024, 256, 0, stream>>>(qkv_w,  qkvw16,  L_ * 3 * D_ * D_);
  cvt_f32_f16<<<512,  256, 0, stream>>>(proj_w, projw16, L_ * D_ * D_);
  cvt_f32_f16<<<1024, 256, 0, stream>>>(ffn_w1, f1w16,   L_ * HID_ * D_);
  cvt_f32_f16<<<1024, 256, 0, stream>>>(ffn_w2, f2w16,   L_ * D_ * HID_);

  embed_kernel<<<M_, 256, 0, stream>>>(W_emb, pos_emb, ids, x);

  const dim3 blk(256);
  for (int i = 0; i < L_; ++i) {
    // h = LN1(x) -> f16
    layernorm_f16<<<M_, 256, 0, stream>>>(x, norm1_g + i * D_,
                                          norm1_b + i * D_, h16);
    // qkv = h @ qkv_w^T + qkv_b   (f32 out for attention)
    gemm_wmma_f16<<<dim3(3 * D_ / 256, M_ / 64), blk, 0, stream>>>(
        h16, qkvw16 + (size_t)i * 3 * D_ * D_, qkv_b + i * 3 * D_, nullptr,
        qkvf, nullptr, 0, 0, M_, 3 * D_, D_);
    // causal softmax attention -> f16 directly
    attn_kernel<<<(B_ * H_ * S_) / 8, 256, 0, stream>>>(qkvf, attn16);
    // x = x + attn @ proj_w^T + proj_b
    gemm_wmma_f16<<<dim3(D_ / 256, M_ / 64), blk, 0, stream>>>(
        attn16, projw16 + (size_t)i * D_ * D_, proj_b + i * D_, x,
        x, nullptr, 0, 0, M_, D_, D_);
    // h = LN2(x) -> f16
    layernorm_f16<<<M_, 256, 0, stream>>>(x, norm2_g + i * D_,
                                          norm2_b + i * D_, h16);
    // hid16 = gelu(h @ ffn_w1^T + b1)  (fused GELU epilogue, f16 out only)
    gemm_wmma_f16<<<dim3(HID_ / 256, M_ / 64), blk, 0, stream>>>(
        h16, f1w16 + (size_t)i * HID_ * D_, ffn_b1 + i * HID_, nullptr,
        nullptr, hid16, 1, 0, M_, HID_, D_);
    // x = x + hid16 @ ffn_w2^T + b2
    gemm_wmma_f16<<<dim3(D_ / 256, M_ / 64), blk, 0, stream>>>(
        hid16, f2w16 + (size_t)i * D_ * HID_, ffn_b2 + i * D_, x,
        x, nullptr, 0, 0, M_, D_, HID_);
  }

  // final LN + logits = LN(x) @ W_emb^T  (262MB write-once -> non-temporal)
  layernorm_f16<<<M_, 256, 0, stream>>>(x, fin_g, fin_b, h16);
  gemm_wmma_f16<<<dim3(V_ / 256, M_ / 64), blk, 0, stream>>>(
      h16, wemb16, nullptr, nullptr, (float*)d_out, nullptr, 0, 1,
      M_, V_, D_);
}
